// MultiHeadedAttention_58531814310504
// MI455X (gfx1250) — matrix operand order repaired and checked by reading, not compiled, not run on hardware
//
#include <hip/hip_runtime.h>
#include <hip/hip_bf16.h>
#include <stdint.h>

typedef __attribute__((ext_vector_type(16))) __bf16 v16bf;
typedef __attribute__((ext_vector_type(8)))  __bf16 v8bf;
typedef __attribute__((ext_vector_type(4)))  __bf16 v4bf;
typedef __attribute__((ext_vector_type(8)))  float  v8f;
typedef __attribute__((ext_vector_type(4)))  unsigned int v4u;
typedef __attribute__((ext_vector_type(8)))  int    v8i;
typedef __attribute__((ext_vector_type(4)))  int    v4i;

#define S_DIM 1024
#define D_DIM 1024
#define H_DIM 16
#define DK_DIM 64

// ---------- helpers ----------

__device__ inline __bf16 f2bf(float f) {
  unsigned u = __builtin_bit_cast(unsigned, f);
  unsigned r = (u + 0x7FFFu + ((u >> 16) & 1u)) >> 16;
  unsigned short s = (unsigned short)r;
  return __builtin_bit_cast(__bf16, s);
}

__device__ inline unsigned lds_off(const void* p) {
  return (unsigned)(unsigned long long)p;  // LDS offset lives in the low 32 bits
}

// Tensor Data Mover: 2D tile of 2-byte elements -> LDS. D# packed per ISA 8.3/8.4.
__device__ inline void tdm_load_2d(unsigned lds_byte, const void* gptr,
                                   unsigned tdim0, unsigned tdim1,
                                   unsigned long long stride0,
                                   unsigned tile0, unsigned tile1) {
  unsigned long long ga = (unsigned long long)gptr;
  v4u g0;
  g0[0] = 1u;                                                 // count=1
  g0[1] = lds_byte;                                           // lds_addr
  g0[2] = (unsigned)ga;                                       // global_addr lo
  g0[3] = (unsigned)((ga >> 32) & 0x1FFFFFFull) | (2u << 30); // addr hi | type=2
  v8i g1;
  g1[0] = (int)(1u << 16);                                    // data_size = 2B
  g1[1] = (int)((tdim0 & 0xFFFFu) << 16);
  g1[2] = (int)((tdim0 >> 16) | ((tdim1 & 0xFFFFu) << 16));
  g1[3] = (int)((tdim1 >> 16) | ((tile0 & 0xFFFFu) << 16));
  g1[4] = (int)(tile1 & 0xFFFFu);
  g1[5] = (int)(unsigned)(stride0 & 0xFFFFFFFFull);
  g1[6] = (int)(unsigned)((stride0 >> 32) & 0xFFFFull);
  g1[7] = 0;
  v4i z4 = {0, 0, 0, 0};
  v8i z8 = {0, 0, 0, 0, 0, 0, 0, 0};
  __builtin_amdgcn_tensor_load_to_lds(g0, g1, z4, z4, z8, 0);
}

__device__ inline v8f wmma_bf16(v16bf a, v16bf b, v8f c) {
  return __builtin_amdgcn_wmma_f32_16x16x32_bf16(false, a, false, b, (short)0, c,
                                                 false, false);
}

// A-fragment 16x32 bf16: per lane the two K-halves are 16B-contiguous ->
// exactly two b128 loads.
__device__ inline v16bf load_a_frag(const __bf16* base, int pitch) {
  const int lane = threadIdx.x & 31;
  const int row = lane & 15;
  const int kb = (lane >> 4) << 3;  // 0 or 8
  const __bf16* p = base + (size_t)row * pitch + kb;
  v8bf lo = *(const v8bf*)(p);
  v8bf hi = *(const v8bf*)(p + 16);
  return __builtin_shufflevector(lo, hi, 0, 1, 2, 3, 4, 5, 6, 7,
                                 8, 9, 10, 11, 12, 13, 14, 15);
}

// B-fragment 32x16 bf16 from row-major KxN tile: lane -> N column, elements -> K (table order).
__device__ inline v16bf load_b_frag(const __bf16* base, int pitch) {
  const int lane = threadIdx.x & 31;
  const int n = lane & 15;
  const int kb = (lane >> 4) << 3;
  v16bf r;
#pragma unroll
  for (int s = 0; s < 8; ++s) {
    r[s] = base[(size_t)(kb + s) * pitch + n];
    r[s + 8] = base[(size_t)(16 + kb + s) * pitch + n];
  }
  return r;
}

// ---------- kernels ----------

__global__ void cvt_f32_to_bf16(const float* __restrict__ x,
                                __bf16* __restrict__ y, size_t n4) {
  size_t i = (size_t)blockIdx.x * blockDim.x + threadIdx.x;
  const size_t st = (size_t)gridDim.x * blockDim.x;
  const float4* x4 = (const float4*)x;
  for (; i < n4; i += st) {
    float4 v = x4[i];
    v4bf o;
    o[0] = f2bf(v.x); o[1] = f2bf(v.y); o[2] = f2bf(v.z); o[3] = f2bf(v.w);
    *(v4bf*)(y + 4 * i) = o;  // single 8-byte store
  }
}

// Fused convert + transpose: W[N][K] f32 -> Wt[K][N] bf16 (so GEMM B-tiles are
// K-major; B-fragments are read from LDS one element per k row).
__global__ __launch_bounds__(256) void transpose_f32_to_bf16(
    const float* __restrict__ W, __bf16* __restrict__ Wt, int N, int K) {
  __shared__ float tile[32][33];
  const int n0 = blockIdx.x * 32, k0 = blockIdx.y * 32;
  const int tx = threadIdx.x & 31, ty = threadIdx.x >> 5;
#pragma unroll
  for (int r = ty; r < 32; r += 8)
    tile[r][tx] = W[(size_t)(n0 + r) * K + k0 + tx];
  __syncthreads();
#pragma unroll
  for (int r = ty; r < 32; r += 8)
    Wt[(size_t)(k0 + r) * N + n0 + tx] = f2bf(tile[tx][r]);
}

// C[M,N] = A[M,K] * Wt[K,N] + bias (Wt pre-transposed, = original W^T).
// layout: 0=f32 [M,N], 1=bf16 [M,N], 2=bf16 [B,H,S,dk], 3=bf16 [B,H,dk,S].
__global__ __launch_bounds__(256) void gemm_wmma_bf16(
    const __bf16* __restrict__ A, const __bf16* __restrict__ Wt,
    const float* __restrict__ bias, void* __restrict__ Cout,
    int M, int N, int K, int layout, int S, int dk) {
  __shared__ __align__(16) __bf16 sA[2][128 * 64];  // 128 rows x 64 K
  __shared__ __align__(16) __bf16 sW[2][64 * 64];   // 64 K x 64 N
  const int m0 = blockIdx.y * 128;
  const int n0 = blockIdx.x * 64;
  const int wave = threadIdx.x >> 5;
  const int lane = threadIdx.x & 31;
  v8f zf = {0.f, 0.f, 0.f, 0.f, 0.f, 0.f, 0.f, 0.f};
  v8f acc[4] = {zf, zf, zf, zf};

  const int nk = K / 64;
  if (wave == 0) {
    tdm_load_2d(lds_off(&sA[0][0]), A + (size_t)m0 * K, (unsigned)K, (unsigned)M,
                (unsigned long long)K, 64u, 128u);
    tdm_load_2d(lds_off(&sW[0][0]), Wt + n0, (unsigned)N, (unsigned)K,
                (unsigned long long)N, 64u, 64u);
    __builtin_amdgcn_s_wait_tensorcnt(0);
  }
  __syncthreads();

  for (int kt = 0; kt < nk; ++kt) {
    const int cur = kt & 1;
    if (wave == 0 && (kt + 1) < nk) {
      const size_t k1 = (size_t)(kt + 1) * 64;  // prefetch next K-slab
      tdm_load_2d(lds_off(&sA[cur ^ 1][0]), A + (size_t)m0 * K + k1, (unsigned)K,
                  (unsigned)M, (unsigned long long)K, 64u, 128u);
      tdm_load_2d(lds_off(&sW[cur ^ 1][0]), Wt + k1 * N + n0, (unsigned)N,
                  (unsigned)K, (unsigned long long)N, 64u, 64u);
    }
    v16bf a0 = load_a_frag(&sA[cur][wave * 16 * 64], 64);
    v16bf a1 = load_a_frag(&sA[cur][wave * 16 * 64 + 32], 64);
#pragma unroll
    for (int nt = 0; nt < 4; ++nt) {
      v16bf b0 = load_b_frag(&sW[cur][nt * 16], 64);
      v16bf b1 = load_b_frag(&sW[cur][32 * 64 + nt * 16], 64);
      acc[nt] = wmma_bf16(a0, b0, acc[nt]);
      acc[nt] = wmma_bf16(a1, b1, acc[nt]);
    }
    if (wave == 0 && (kt + 1) < nk) __builtin_amdgcn_s_wait_tensorcnt(0);
    __syncthreads();
  }

  // Epilogue: C-layout = VGPR r, lanes 0-15 -> M=r, lanes 16-31 -> M=8+r.
  const int col = lane & 15, hi = lane >> 4;
  const int H = N / dk;
#pragma unroll
  for (int nt = 0; nt < 4; ++nt) {
#pragma unroll
    for (int r = 0; r < 8; ++r) {
      const int row = m0 + wave * 16 + r + 8 * hi;
      const int cn = n0 + nt * 16 + col;
      const float v = acc[nt][r] + bias[cn];
      if (layout == 0) {
        ((float*)Cout)[(size_t)row * N + cn] = v;
      } else if (layout == 1) {
        ((__bf16*)Cout)[(size_t)row * N + cn] = f2bf(v);
      } else {
        const int bb = row / S, s = row % S;
        const int h = cn / dk, d = cn % dk;
        const size_t idx = (layout == 2)
                               ? ((((size_t)bb * H + h) * S + s) * dk + d)
                               : ((((size_t)bb * H + h) * dk + d) * S + s);
        ((__bf16*)Cout)[idx] = f2bf(v);
      }
    }
  }
}

// Fused flash attention. Q:[B,H,S,dk] bf16, Kt:[B,H,dk,S] bf16, V:[B,H,S,dk] bf16.
// O (bf16, [B,S,D]) = softmax(QK^T/sqrt(2dk) + pos, mask) @ V, per (b,h,q-tile).
__global__ __launch_bounds__(256) void flash_attn_wmma(
    const __bf16* __restrict__ Q, const __bf16* __restrict__ Kt,
    const __bf16* __restrict__ V, const float* __restrict__ pos,
    const int* __restrict__ mask, __bf16* __restrict__ Oout,
    int B, int H, int S, int dk) {
  __shared__ __align__(16) __bf16 sKt[2][64 * 64];  // dk x kv
  __shared__ __align__(16) __bf16 sV[2][64 * 64];   // kv x dk
  __shared__ __align__(16) __bf16 sP[8][16 * 64];   // per-wave P staging

  const int b = blockIdx.z, h = blockIdx.y;
  const int q0 = blockIdx.x * 128;
  const int wave = threadIdx.x >> 5;
  const int lane = threadIdx.x & 31;
  const int qw = q0 + wave * 16;
  const int D = H * dk;

  const __bf16* Qb  = Q  + (((size_t)b * H + h) * S) * dk;
  const __bf16* Ktb = Kt + (((size_t)b * H + h) * dk) * S;
  const __bf16* Vb  = V  + (((size_t)b * H + h) * S) * dk;
  const float*  posb = pos + ((size_t)h * S) * S;
  const int*    maskb = mask + ((size_t)b * S) * S;

  // Q fragments: 16 rows x dk=64 -> two K=32 A-frags, loaded once (b128 global).
  v16bf qa0 = load_a_frag(Qb + (size_t)qw * dk, dk);
  v16bf qa1 = load_a_frag(Qb + (size_t)qw * dk + 32, dk);

  v8f zf = {0.f, 0.f, 0.f, 0.f, 0.f, 0.f, 0.f, 0.f};
  v8f o[4] = {zf, zf, zf, zf};
  float m_r[8], l_r[8];
#pragma unroll
  for (int r = 0; r < 8; ++r) { m_r[r] = -1e30f; l_r[r] = 0.f; }

  const float scale = 0.08838834764831845f;  // 1/sqrt(2*dk) = 1/sqrt(128)
  const int nkv = S / 64;

  if (wave == 0) {
    tdm_load_2d(lds_off(&sKt[0][0]), Ktb, (unsigned)S, 64u,
                (unsigned long long)S, 64u, 64u);
    tdm_load_2d(lds_off(&sV[0][0]), Vb, 64u, (unsigned)S, 64ull, 64u, 64u);
    __builtin_amdgcn_s_wait_tensorcnt(0);
  }
  __syncthreads();

  for (int kb = 0; kb < nkv; ++kb) {
    const int cur = kb & 1;
    const int kv0 = kb * 64;
    if (wave == 0 && (kb + 1) < nkv) {
      const int kv1 = kv0 + 64;
      tdm_load_2d(lds_off(&sKt[cur ^ 1][0]), Ktb + kv1, (unsigned)S, 64u,
                  (unsigned long long)S, 64u, 64u);
      tdm_load_2d(lds_off(&sV[cur ^ 1][0]), Vb + (size_t)kv1 * dk, 64u,
                  (unsigned)S, 64ull, 64u, 64u);
    }
    const __bf16* tK = &sKt[cur][0];
    const __bf16* tV = &sV[cur][0];
    const int col = lane & 15, hi = lane >> 4;

    // scores S = Q * Kt tile (16 x 64 per wave), + bias/mask in-register
    v8f s4[4];
#pragma unroll
    for (int nt = 0; nt < 4; ++nt) {
      v8f c = zf;
      v16bf b0 = load_b_frag(tK + nt * 16, 64);
      v16bf b1 = load_b_frag(tK + 32 * 64 + nt * 16, 64);
      c = wmma_bf16(qa0, b0, c);
      c = wmma_bf16(qa1, b1, c);
      const int kvcol = kv0 + nt * 16 + col;
#pragma unroll
      for (int r = 0; r < 8; ++r) {
        const int qr = qw + r + 8 * hi;
        float sc = c[r] * scale + posb[(size_t)qr * S + kvcol];
        if (maskb[(size_t)qr * S + kvcol] == 0) sc = -1e9f;
        c[r] = sc;
      }
      s4[nt] = c;
    }
    if ((kb + 1) < nkv)  // keep next bias tile warm
      __builtin_prefetch(&posb[(size_t)qw * S + kv0 + 64], 0, 1);

    // online softmax: row max/sum via shfl_xor within each 16-lane half
    float nm[8];
#pragma unroll
    for (int r = 0; r < 8; ++r) {
      float rm = fmaxf(fmaxf(s4[0][r], s4[1][r]), fmaxf(s4[2][r], s4[3][r]));
      for (int mm = 1; mm < 16; mm <<= 1)
        rm = fmaxf(rm, __shfl_xor(rm, mm, 32));
      nm[r] = fmaxf(m_r[r], rm);
    }
    float alpha[8], rs[8];
#pragma unroll
    for (int r = 0; r < 8; ++r) {
      alpha[r] = __expf(m_r[r] - nm[r]);
      m_r[r] = nm[r];
      rs[r] = 0.f;
    }
    __bf16* pw = &sP[wave][0];
#pragma unroll
    for (int nt = 0; nt < 4; ++nt) {
#pragma unroll
      for (int r = 0; r < 8; ++r) {
        const float p = __expf(s4[nt][r] - m_r[r]);
        rs[r] += p;
        pw[(size_t)(r + 8 * hi) * 64 + nt * 16 + col] = f2bf(p);
      }
    }
#pragma unroll
    for (int r = 0; r < 8; ++r) {
      for (int mm = 1; mm < 16; mm <<= 1) rs[r] += __shfl_xor(rs[r], mm, 32);
      l_r[r] = l_r[r] * alpha[r] + rs[r];
    }
#pragma unroll
    for (int d = 0; d < 4; ++d)
#pragma unroll
      for (int r = 0; r < 8; ++r) o[d][r] = o[d][r] * alpha[r];

    // P staged through LDS to re-enter as A-fragments (wave-local)
    asm volatile("s_wait_dscnt 0" ::: "memory");
    v16bf pa0 = load_a_frag(pw, 64);
    v16bf pa1 = load_a_frag(pw + 32, 64);
#pragma unroll
    for (int d = 0; d < 4; ++d) {
      v16bf bv0 = load_b_frag(tV + d * 16, 64);
      v16bf bv1 = load_b_frag(tV + 32 * 64 + d * 16, 64);
      o[d] = wmma_bf16(pa0, bv0, o[d]);
      o[d] = wmma_bf16(pa1, bv1, o[d]);
    }

    if (wave == 0 && (kb + 1) < nkv) __builtin_amdgcn_s_wait_tensorcnt(0);
    __syncthreads();
  }

  // finalize: divide by l, scatter back to [B,S,D] bf16
  const int col = lane & 15, hi = lane >> 4;
#pragma unroll
  for (int d = 0; d < 4; ++d) {
#pragma unroll
    for (int r = 0; r < 8; ++r) {
      const int qr = qw + r + 8 * hi;
      const int dc = h * dk + d * 16 + col;
      Oout[((size_t)b * S + qr) * D + dc] = f2bf(o[d][r] / l_r[r]);
    }
  }
}

// ---------- host launcher ----------

extern "C" void kernel_launch(void* const* d_in, const int* in_sizes, int n_in,
                              void* d_out, int out_size, void* d_ws, size_t ws_size,
                              hipStream_t stream) {
  const float* query = (const float*)d_in[0];
  const float* key_  = (const float*)d_in[1];
  const float* value = (const float*)d_in[2];
  const int*   mask  = (const int*)d_in[3];
  const float* pos   = (const float*)d_in[4];
  const float* wq = (const float*)d_in[5];
  const float* bq = (const float*)d_in[6];
  const float* wk = (const float*)d_in[7];
  const float* bk = (const float*)d_in[8];
  const float* wv = (const float*)d_in[9];
  const float* bv = (const float*)d_in[10];
  const float* wo = (const float*)d_in[11];
  const float* bo = (const float*)d_in[12];

  const int S = S_DIM, D = D_DIM, H = H_DIM, dk = DK_DIM;
  const int B = in_sizes[0] / (S * D);
  const int M = B * S;

  char* ws = (char*)d_ws;
  size_t off = 0;
  auto alloc = [&](size_t bytes) -> void* {
    void* p = ws + off;
    off += (bytes + 255) & ~(size_t)255;
    return p;
  };
  __bf16* Xq  = (__bf16*)alloc((size_t)M * D * 2);
  __bf16* Xk  = (__bf16*)alloc((size_t)M * D * 2);
  __bf16* Xv  = (__bf16*)alloc((size_t)M * D * 2);
  __bf16* Wqt = (__bf16*)alloc((size_t)D * D * 2);
  __bf16* Wkt = (__bf16*)alloc((size_t)D * D * 2);
  __bf16* Wvt = (__bf16*)alloc((size_t)D * D * 2);
  __bf16* Wot = (__bf16*)alloc((size_t)D * D * 2);
  __bf16* Qh  = (__bf16*)alloc((size_t)M * D * 2);   // [B,H,S,dk]
  __bf16* Kth = (__bf16*)alloc((size_t)M * D * 2);   // [B,H,dk,S]
  __bf16* Vh  = (__bf16*)alloc((size_t)M * D * 2);   // [B,H,S,dk]
  __bf16* Att = (__bf16*)alloc((size_t)M * D * 2);   // [B,S,D]
  __bf16* X1  = (__bf16*)alloc((size_t)M * D * 2);   // [B,S,D]
  (void)ws_size; (void)n_in; (void)out_size;

  // 1) stage activations bf16; weights fused convert+transpose (W^T, K-major)
  cvt_f32_to_bf16<<<1024, 256, 0, stream>>>(query, Xq, (size_t)M * D / 4);
  cvt_f32_to_bf16<<<1024, 256, 0, stream>>>(key_,  Xk, (size_t)M * D / 4);
  cvt_f32_to_bf16<<<1024, 256, 0, stream>>>(value, Xv, (size_t)M * D / 4);
  dim3 tgrid(D / 32, D / 32);
  transpose_f32_to_bf16<<<tgrid, 256, 0, stream>>>(wq, Wqt, D, D);
  transpose_f32_to_bf16<<<tgrid, 256, 0, stream>>>(wk, Wkt, D, D);
  transpose_f32_to_bf16<<<tgrid, 256, 0, stream>>>(wv, Wvt, D, D);
  transpose_f32_to_bf16<<<tgrid, 256, 0, stream>>>(wo, Wot, D, D);

  // 2) projections (K written pre-transposed per head)
  dim3 ggrid(D / 64, M / 128);
  gemm_wmma_bf16<<<ggrid, 256, 0, stream>>>(Xq, Wqt, bq, Qh,  M, D, D, 2, S, dk);
  gemm_wmma_bf16<<<ggrid, 256, 0, stream>>>(Xk, Wkt, bk, Kth, M, D, D, 3, S, dk);
  gemm_wmma_bf16<<<ggrid, 256, 0, stream>>>(Xv, Wvt, bv, Vh,  M, D, D, 2, S, dk);

  // 3) fused flash attention
  dim3 fgrid(S / 128, H, B);
  flash_attn_wmma<<<fgrid, 256, 0, stream>>>(Qh, Kth, Vh, pos, mask, Att,
                                             B, H, S, dk);

  // 4) output projection applied twice (faithful to reference)
  gemm_wmma_bf16<<<ggrid, 256, 0, stream>>>(Att, Wot, bo, X1, M, D, D, 1, S, dk);
  gemm_wmma_bf16<<<ggrid, 256, 0, stream>>>(X1, Wot, bo, d_out, M, D, D, 0, S, dk);
}
